// Traget_surround_38732015075684
// MI455X (gfx1250) — matrix operand order repaired and checked by reading, not compiled, not run on hardware
//
#include <hip/hip_runtime.h>
#include <hip/hip_bf16.h>

typedef __attribute__((ext_vector_type(16))) __bf16 v16bf;
typedef __attribute__((ext_vector_type(8)))  __bf16 v8bf;
typedef __attribute__((ext_vector_type(8)))  float  v8f;

#define NEG_SLOPE 0.01f
#define B_   128
#define N_   2048
#define D_   128
#define H_   128
#define LDST 136   // LDS row stride in bf16 elements (136*2=272 B, 16B-aligned, avoids bank conflicts)

__device__ __forceinline__ float leaky(float x) {
    return x >= 0.0f ? x : NEG_SLOPE * x;
}

// ---------------------------------------------------------------------------
// Kernel 1: t1[b][h] = b1[h] + sum_d target[b][d] * W1[d][h]   (top half of W1)
// ---------------------------------------------------------------------------
__global__ __launch_bounds__(128) void t1_kernel(const float* __restrict__ target,
                                                 const float* __restrict__ W1,
                                                 const float* __restrict__ b1,
                                                 float* __restrict__ t1) {
    int b = blockIdx.x, h = threadIdx.x;
    const float* tg = target + b * D_;
    float acc = b1[h];
    #pragma unroll 4
    for (int d = 0; d < D_; ++d) acc += tg[d] * W1[d * H_ + h];
    t1[b * H_ + h] = acc;
}

// ---------------------------------------------------------------------------
// Kernel 2: scores[b][n] = leaky(t1[b] + surr[b,n,:] @ W1_bot) . W2
// bf16 WMMA main GEMM. Block = (b, 128 n-rows); wave = 16 n-rows x all H.
// ---------------------------------------------------------------------------
__global__ __launch_bounds__(256) void score_kernel(const float* __restrict__ surr,
                                                    const float* __restrict__ W1,
                                                    const float* __restrict__ t1,
                                                    const float* __restrict__ W2,
                                                    float* __restrict__ scores) {
    __shared__ __bf16 w1b[H_ * LDST];   // W1_bot transposed: w1b[h*LDST + k], bf16

    const int tid = threadIdx.x;

    // Stage W1 rows 128..255 (the "surrounding" half), transposed, as bf16.
    // Global reads coalesced over h; 16384 elems / 256 threads = 64 iters.
    #pragma unroll 4
    for (int i = 0; i < 64; ++i) {
        int idx = i * 256 + tid;
        int k = idx >> 7, h = idx & 127;
        w1b[h * LDST + k] = (__bf16)W1[(D_ + k) * H_ + h];
    }
    __syncthreads();

    const int b    = blockIdx.x >> 4;   // 128 batches
    const int nblk = blockIdx.x & 15;   // 16 blocks of 128 rows
    const int wave = tid >> 5;
    const int lane = tid & 31;
    const int m    = lane & 15;         // row within 16-row tile / column within h-tile
    const int half = lane >> 4;
    const int n0   = nblk * 128 + wave * 16;

    // --- A fragments: 16x32 bf16 per K-chunk, ISA lane layout ---
    // lane (m, half): elems 0..7  = K in [32c + 8*half,      +8)
    //                 elems 8..15 = K in [32c + 16 + 8*half, +8)
    const float4* rowv = (const float4*)(surr + ((size_t)(b * N_ + n0 + m)) * D_);
    v16bf afrag[4];
    #pragma unroll
    for (int c = 0; c < 4; ++c) {
        const int q0 = (c * 32 + half * 8) >> 2;       // float4 index
        float4 f0 = rowv[q0];
        float4 f1 = rowv[q0 + 1];
        float4 f2 = rowv[q0 + 4];
        float4 f3 = rowv[q0 + 5];
        v16bf a;
        a[0]  = (__bf16)f0.x; a[1]  = (__bf16)f0.y; a[2]  = (__bf16)f0.z; a[3]  = (__bf16)f0.w;
        a[4]  = (__bf16)f1.x; a[5]  = (__bf16)f1.y; a[6]  = (__bf16)f1.z; a[7]  = (__bf16)f1.w;
        a[8]  = (__bf16)f2.x; a[9]  = (__bf16)f2.y; a[10] = (__bf16)f2.z; a[11] = (__bf16)f2.w;
        a[12] = (__bf16)f3.x; a[13] = (__bf16)f3.y; a[14] = (__bf16)f3.z; a[15] = (__bf16)f3.w;
        afrag[c] = a;
    }

    float sacc[8] = {0.f, 0.f, 0.f, 0.f, 0.f, 0.f, 0.f, 0.f};

    #pragma unroll
    for (int ht = 0; ht < 8; ++ht) {
        const int h0 = ht * 16 + m;     // this lane's h-column
        v8f acc = {};
        #pragma unroll
        for (int c = 0; c < 4; ++c) {
            // B fragment 32x16: lane holds column n=m; elems 0..7 = K in [32c+8*half, +8), 8..15 = K in [32c+16+8*half, +8)
            const v8bf* bp = (const v8bf*)&w1b[h0 * LDST + c * 32 + half * 8];
            v8bf blo = bp[0];
            v8bf bhi = bp[2];
            v16bf bf = __builtin_shufflevector(blo, bhi,
                0, 1, 2, 3, 4, 5, 6, 7, 8, 9, 10, 11, 12, 13, 14, 15);
            acc = __builtin_amdgcn_wmma_f32_16x16x32_bf16(
                false, afrag[c], false, bf, (short)0, acc, false, false);
        }
        const float t  = t1[b * H_ + h0];
        const float w2 = W2[h0];
        #pragma unroll
        for (int v = 0; v < 8; ++v) {
            // D layout: lane col = h0, VGPR v = n-row (v + 8*half)
            sacc[v] += leaky(acc[v] + t) * w2;
        }
    }

    // Reduce over the 16 lanes of each half (masks 1,2,4,8 stay within a half)
    #pragma unroll
    for (int v = 0; v < 8; ++v) {
        float s = sacc[v];
        s += __shfl_xor(s, 1, 32);
        s += __shfl_xor(s, 2, 32);
        s += __shfl_xor(s, 4, 32);
        s += __shfl_xor(s, 8, 32);
        sacc[v] = s;
    }
    if (m == 0) {
        #pragma unroll
        for (int v = 0; v < 8; ++v)
            scores[b * N_ + n0 + 8 * half + v] = sacc[v];
    }
}

// ---------------------------------------------------------------------------
// Kernel 3: weights[b] = softmax(scores[b])   (b2 cancels: softmax shift-invariant)
// ---------------------------------------------------------------------------
__global__ __launch_bounds__(256) void softmax_kernel(const float* __restrict__ scores,
                                                      float* __restrict__ weights) {
    __shared__ float red_m[8];
    __shared__ float red_s[8];
    const int b = blockIdx.x, tid = threadIdx.x;
    const int wave = tid >> 5, lane = tid & 31;
    const float* s = scores + b * N_;

    float v[8];
    float mx = -3.402823466e+38f;
    #pragma unroll
    for (int i = 0; i < 8; ++i) { v[i] = s[tid + i * 256]; mx = fmaxf(mx, v[i]); }
    #pragma unroll
    for (int off = 16; off >= 1; off >>= 1) mx = fmaxf(mx, __shfl_xor(mx, off, 32));
    if (lane == 0) red_m[wave] = mx;
    __syncthreads();
    float gmx = red_m[0];
    #pragma unroll
    for (int j = 1; j < 8; ++j) gmx = fmaxf(gmx, red_m[j]);

    float sum = 0.f;
    #pragma unroll
    for (int i = 0; i < 8; ++i) { v[i] = __expf(v[i] - gmx); sum += v[i]; }
    #pragma unroll
    for (int off = 16; off >= 1; off >>= 1) sum += __shfl_xor(sum, off, 32);
    if (lane == 0) red_s[wave] = sum;
    __syncthreads();
    float gsum = 0.f;
    #pragma unroll
    for (int j = 0; j < 8; ++j) gsum += red_s[j];

    const float inv = 1.0f / gsum;
    #pragma unroll
    for (int i = 0; i < 8; ++i) weights[b * N_ + tid + i * 256] = v[i] * inv;
}

// ---------------------------------------------------------------------------
// Kernel 4: attended[b][d] = sum_n weights[b][n] * surr[b][n][d]
// ---------------------------------------------------------------------------
__global__ __launch_bounds__(512) void attended_kernel(const float* __restrict__ surr,
                                                       const float* __restrict__ weights,
                                                       float* __restrict__ attended) {
    __shared__ float part[4][D_];
    const int b = blockIdx.x, tid = threadIdx.x;
    const int d = tid & 127, c = tid >> 7;            // 4 chunks of 512 n
    const float* w  = weights + b * N_;
    const float* sb = surr + (size_t)b * N_ * D_;
    float acc = 0.f;
    #pragma unroll 4
    for (int n = c * 512; n < c * 512 + 512; ++n)
        acc += w[n] * sb[(size_t)n * D_ + d];
    part[c][d] = acc;
    __syncthreads();
    if (c == 0)
        attended[b * D_ + d] = part[0][d] + part[1][d] + part[2][d] + part[3][d];
}

// ---------------------------------------------------------------------------
// Kernel 5: out[b] = leaky([target[b] | attended[b]] @ W3 + b3)
// ---------------------------------------------------------------------------
__global__ __launch_bounds__(128) void out_kernel(const float* __restrict__ target,
                                                  const float* __restrict__ attended,
                                                  const float* __restrict__ W3,
                                                  const float* __restrict__ b3,
                                                  float* __restrict__ out) {
    const int b = blockIdx.x, d = threadIdx.x;
    const float* tg = target + b * D_;
    const float* at = attended + b * D_;
    float acc = b3[d];
    #pragma unroll 4
    for (int k = 0; k < D_; ++k) acc += tg[k] * W3[k * D_ + d];
    #pragma unroll 4
    for (int k = 0; k < D_; ++k) acc += at[k] * W3[(D_ + k) * D_ + d];
    out[b * D_ + d] = leaky(acc);
}

// ---------------------------------------------------------------------------
extern "C" void kernel_launch(void* const* d_in, const int* in_sizes, int n_in,
                              void* d_out, int out_size, void* d_ws, size_t ws_size,
                              hipStream_t stream) {
    const float* target = (const float*)d_in[0];   // (128,128)
    const float* surr   = (const float*)d_in[1];   // (128,2048,128)
    const float* W1     = (const float*)d_in[2];   // (256,128)
    const float* b1     = (const float*)d_in[3];   // (128,)
    const float* W2     = (const float*)d_in[4];   // (128,1)
    const float* b2     = (const float*)d_in[5];   // (1,)  -- cancels in softmax
    const float* W3     = (const float*)d_in[6];   // (256,128)
    const float* b3     = (const float*)d_in[7];   // (128,)
    (void)b2; (void)in_sizes; (void)n_in; (void)out_size; (void)ws_size;

    float* ws       = (float*)d_ws;
    float* t1       = ws;                                   // 128*128
    float* scores   = t1 + B_ * H_;                          // 128*2048
    float* weights  = scores + B_ * N_;                      // 128*2048
    float* attended = weights + B_ * N_;                     // 128*128

    t1_kernel      <<<B_,        128, 0, stream>>>(target, W1, b1, t1);
    score_kernel   <<<B_ * 16,   256, 0, stream>>>(surr, W1, t1, W2, scores);
    softmax_kernel <<<B_,        256, 0, stream>>>(scores, weights);
    attended_kernel<<<B_,        512, 0, stream>>>(surr, weights, attended);
    out_kernel     <<<B_,        128, 0, stream>>>(target, attended, W3, b3, (float*)d_out);
}
